// DecoderGroupedQueryHeadAttentionRope_80736795231129
// MI455X (gfx1250) — matrix operand order repaired and checked by reading, not compiled, not run on hardware
//
#include <hip/hip_runtime.h>
#include <hip/hip_bf16.h>

typedef __attribute__((ext_vector_type(16))) _Float16 v16h;
typedef __attribute__((ext_vector_type(8)))  _Float16 v8h;
typedef __attribute__((ext_vector_type(8)))  float    v8f;
typedef __attribute__((ext_vector_type(4)))  unsigned int u32x4;
typedef __attribute__((ext_vector_type(8)))  int      i32x8;
typedef __attribute__((ext_vector_type(4)))  int      i32x4;

#define NB    4
#define TT    2048
#define CC    2048
#define NH    16
#define NKV   4
#define HD    128
#define GROUP 4
#define NTOT  (CC + 2 * NKV * HD)   // 3072 fused output columns (Q | K | V)

#if defined(__gfx1250__) && __has_builtin(__builtin_amdgcn_tensor_load_to_lds) && \
    __has_builtin(__builtin_amdgcn_s_wait_tensorcnt)
#define HAVE_TDM 1
#else
#define HAVE_TDM 0
#endif

// ---------------------------------------------------------------------------
// WMMA helpers (wave32, 16x16x32 f16 -> f32 accumulate)
// ---------------------------------------------------------------------------
__device__ __forceinline__ v16h cat16(v8h lo, v8h hi) {
  return __builtin_shufflevector(lo, hi, 0, 1, 2, 3, 4, 5, 6, 7,
                                         8, 9, 10, 11, 12, 13, 14, 15);
}

// A-matrix 16x32 fragment (ISA 7.12.2): lane l -> row = l%16, half = l/16.
// half 0: K {0..7, 16..23}; half 1: K {8..15, 24..31}. Two 16B LDS loads.
__device__ __forceinline__ v16h ldsA_frag(const _Float16* base, int stride,
                                          int row, int kofs, int lane) {
  const _Float16* p = base + (size_t)(row + (lane & 15)) * stride + kofs + ((lane >> 4) << 3);
  v8h lo = *(const v8h*)p;
  v8h hi = *(const v8h*)(p + 16);
  return cat16(lo, hi);
}

// B-matrix 32x16 fragment: lane l -> col n = l%16, half = l/16: K {0..7, 16..23} / {8..15, 24..31}.
__device__ __forceinline__ v16h ldsB_frag(const _Float16* base, int stride,
                                          int col, int kofs, int lane) {
  const _Float16* p = base + (size_t)(col + (lane & 15)) * stride + kofs + ((lane >> 4) << 3);
  v8h lo = *(const v8h*)p;
  v8h hi = *(const v8h*)(p + 16);
  return cat16(lo, hi);
}

__device__ __forceinline__ v8f wmma16(v16h a, v16h b, v8f c) {
  return __builtin_amdgcn_wmma_f32_16x16x32_f16(false, a, false, b,
                                                (short)0, c, false, false);
}

// ---------------------------------------------------------------------------
// Tensor Data Mover: 2D f16 tile (rows x 128) global -> LDS, with LDS padding
// producing a 136-half row stride (64 DW row + 4 DW pad). ISA 8.3/8.4.
// ---------------------------------------------------------------------------
#if HAVE_TDM
__device__ __forceinline__ void tdm_load_tile_f16(unsigned lds_off,
                                                  const _Float16* gptr,
                                                  unsigned rows) {
  const unsigned long long ga = (unsigned long long)(const void*)gptr;
  u32x4 g0;
  g0[0] = 1u;                                   // count=1, user descriptor
  g0[1] = lds_off;                              // lds_addr (bytes)
  g0[2] = (unsigned)(ga & 0xffffffffu);         // global_addr[31:0]
  g0[3] = (unsigned)((ga >> 32) & 0x1ffffffu)   // global_addr[56:32]
          | (2u << 30);                         // type = 2 ("image")
  i32x8 g1;
  // data_size=1 (2B), pad_enable, pad_interval=5 (64 DW), pad_amount=3 (4 DW)
  g1[0] = (1 << 16) | (1 << 20) | (5 << 22) | (3 << 25);
  g1[1] = (int)((HD & 0xffff) << 16);           // tensor_dim0[15:0] = 128
  g1[2] = (int)((rows & 0xffffu) << 16);        // tensor_dim0 hi=0 | tensor_dim1 lo
  g1[3] = (int)((HD & 0xffff) << 16);           // tensor_dim1 hi=0 | tile_dim0 = 128
  g1[4] = (int)(rows & 0xffffu);                // tile_dim1 = rows, tile_dim2 = 0
  g1[5] = HD;                                   // tensor_dim0_stride = 128 elems
  g1[6] = 0;
  g1[7] = 0;
  const i32x4 z4 = {0, 0, 0, 0};
#if __has_include(<hip/amd_detail/amd_gfx1250_TDM.h>)
  const i32x8 z8 = {0, 0, 0, 0, 0, 0, 0, 0};
  __builtin_amdgcn_tensor_load_to_lds(g0, g1, z4, z4, z8, 0);
#else
  __builtin_amdgcn_tensor_load_to_lds(g0, g1, z4, z4, 0);
#endif
}
#endif

// ---------------------------------------------------------------------------
// Kernel 1: fused QKV projection. y = x @ [Wq;Wkv]^T, written as f16 into
// head-major Q (B,H,T,HD), K (B,KV,T,HD), V (B,KV,T,HD).
// Block tile 64(M)x64(N), 4 waves of 32x32, K-chunks of 32.
// ---------------------------------------------------------------------------
__global__ __launch_bounds__(128) void k_qkv(const float* __restrict__ x,
                                             const float* __restrict__ Wq,
                                             const float* __restrict__ Wkv,
                                             _Float16* __restrict__ Qo,
                                             _Float16* __restrict__ Ko,
                                             _Float16* __restrict__ Vo) {
  __shared__ __attribute__((aligned(16))) _Float16 sA[64][40];
  __shared__ __attribute__((aligned(16))) _Float16 sB[64][40];
  const int tid = threadIdx.x;
  const int lane = tid & 31;
  const int wave = tid >> 5;
  const int mBase = blockIdx.y * 64;
  const int nBase = blockIdx.x * 64;
  const int wm = (wave >> 1) * 32;
  const int wn = (wave & 1) * 32;
  v8f acc[2][2] = {};
  const int lr = tid >> 3;        // 0..15
  const int lk = (tid & 7) * 4;   // 0..28

  for (int kt = 0; kt < CC; kt += 32) {
#pragma unroll
    for (int rg = 0; rg < 64; rg += 16) {
      const float* xrow = x + (size_t)(mBase + lr + rg) * CC + kt + lk;
      const float4 va = *(const float4*)xrow;
      sA[lr + rg][lk + 0] = (_Float16)va.x;
      sA[lr + rg][lk + 1] = (_Float16)va.y;
      sA[lr + rg][lk + 2] = (_Float16)va.z;
      sA[lr + rg][lk + 3] = (_Float16)va.w;
      const int n = nBase + lr + rg;
      const float* wrow = (n < CC) ? (Wq + (size_t)n * CC)
                                   : (Wkv + (size_t)(n - CC) * CC);
      const float4 vb = *(const float4*)(wrow + kt + lk);
      sB[lr + rg][lk + 0] = (_Float16)vb.x;
      sB[lr + rg][lk + 1] = (_Float16)vb.y;
      sB[lr + rg][lk + 2] = (_Float16)vb.z;
      sB[lr + rg][lk + 3] = (_Float16)vb.w;
      if (kt + 32 < CC) {                       // global_prefetch_b8 next chunk
        __builtin_prefetch(xrow + 32, 0, 0);
        __builtin_prefetch(wrow + kt + 32 + lk, 0, 0);
      }
    }
    __syncthreads();
#pragma unroll
    for (int i = 0; i < 2; ++i) {
      v16h af = ldsA_frag(&sA[0][0], 40, wm + i * 16, 0, lane);
#pragma unroll
      for (int j = 0; j < 2; ++j) {
        v16h bf = ldsB_frag(&sB[0][0], 40, wn + j * 16, 0, lane);
        acc[i][j] = wmma16(af, bf, acc[i][j]);
      }
    }
    __syncthreads();
  }

  // Epilogue: C-layout (row = e + 8*(lane/16), col = lane%16), route to Q/K/V.
  const int rsel = (lane >> 4) * 8;
  const int csel = lane & 15;
#pragma unroll
  for (int i = 0; i < 2; ++i)
#pragma unroll
    for (int j = 0; j < 2; ++j)
#pragma unroll
      for (int e = 0; e < 8; ++e) {
        const int m = mBase + wm + i * 16 + e + rsel;
        const int n = nBase + wn + j * 16 + csel;
        const int bb = m >> 11;       // m / 2048
        const int t = m & 2047;
        const _Float16 val = (_Float16)acc[i][j][e];
        const int d = n & 127;
        if (n < CC) {
          const int h = n >> 7;
          Qo[(((size_t)bb * NH + h) * TT + t) * HD + d] = val;
        } else if (n < CC + NKV * HD) {
          const int h = (n - CC) >> 7;
          Ko[(((size_t)bb * NKV + h) * TT + t) * HD + d] = val;
        } else {
          const int h = (n - CC - NKV * HD) >> 7;
          Vo[(((size_t)bb * NKV + h) * TT + t) * HD + d] = val;
        }
      }
}

// ---------------------------------------------------------------------------
// Kernel 2: rotate-half RoPE applied in place to f16 Q and K.
// ---------------------------------------------------------------------------
__global__ __launch_bounds__(256) void k_rope(_Float16* __restrict__ Qh,
                                              _Float16* __restrict__ Kh,
                                              const float* __restrict__ cs,
                                              const float* __restrict__ sn) {
  long idx = (long)blockIdx.x * 256 + threadIdx.x;
  const long nq = (long)NB * NH * TT * (HD / 2);
  _Float16* p;
  int heads;
  if (idx < nq) { p = Qh; heads = NH; }
  else          { idx -= nq; p = Kh; heads = NKV; }
  const int d = (int)(idx & (HD / 2 - 1));
  const int t = (int)(idx >> 6) & (TT - 1);
  const long r = idx >> 17;                 // T * HD/2 == 2^17
  const int h = (int)(r % heads);
  const int b = (int)(r / heads);
  const size_t base = (((size_t)b * heads + h) * TT + t) * HD;
  const float c = cs[t * (HD / 2) + d];
  const float s = sn[t * (HD / 2) + d];
  const float a0 = (float)p[base + d];
  const float a1 = (float)p[base + (HD / 2) + d];
  p[base + d]            = (_Float16)(a0 * c - a1 * s);
  p[base + (HD / 2) + d] = (_Float16)(a1 * c + a0 * s);
}

// ---------------------------------------------------------------------------
// Kernel 3: causal flash attention. Block = (b, h, 64 q-rows), 4 waves each
// owning 16 q-rows. Q/K tiles DMA'd by the Tensor Data Mover straight into
// padded LDS (stride 136 halves); V transposed through registers. Online
// softmax in fp32 with 16-lane butterflies matching the WMMA C-layout.
// ---------------------------------------------------------------------------
__global__ __launch_bounds__(128) void k_flash(const _Float16* __restrict__ Qh,
                                               const _Float16* __restrict__ Kh,
                                               const _Float16* __restrict__ Vh,
                                               _Float16* __restrict__ Ao) {
  __shared__ __attribute__((aligned(16))) _Float16 sQ[64][136];
  __shared__ __attribute__((aligned(16))) _Float16 sK[32][136];
  __shared__ __attribute__((aligned(16))) _Float16 sVt[128][40];
  __shared__ __attribute__((aligned(16))) _Float16 sP[4][16][40];
  const int tid = threadIdx.x, lane = tid & 31, wave = tid >> 5;
  const int qBase = blockIdx.x * 64;
  const int h = blockIdx.y;
  const int b = blockIdx.z;
  const int hkv = h / GROUP;
  const _Float16* Qp = Qh + (((size_t)b * NH + h) * TT + qBase) * HD;
  const _Float16* Kp = Kh + (((size_t)b * NKV + hkv) * TT) * HD;
  const _Float16* Vp = Vh + (((size_t)b * NKV + hkv) * TT) * HD;

#if HAVE_TDM
  if (wave == 0) {
    tdm_load_tile_f16((unsigned)(unsigned long long)(void*)&sQ[0][0], Qp, 64);
    __builtin_amdgcn_s_wait_tensorcnt((short)0);
  }
#else
  for (int c0 = tid; c0 < 64 * 16; c0 += 128) {
    const int r = c0 >> 4, kc = (c0 & 15) << 3;
    *(v8h*)&sQ[r][kc] = *(const v8h*)(Qp + (size_t)r * HD + kc);
  }
#endif
  __syncthreads();

  v16h aQ[4];
#pragma unroll
  for (int kc = 0; kc < 4; ++kc)
    aQ[kc] = ldsA_frag(&sQ[0][0], 136, wave * 16, kc * 32, lane);

  v8f o[8] = {};
  float rmax[8], rsum[8];
#pragma unroll
  for (int e = 0; e < 8; ++e) { rmax[e] = -1e30f; rsum[e] = 0.0f; }

  const float scale = 0.08838834764831845f;   // 1/sqrt(128)
  const int rsel = (lane >> 4) * 8;
  const int csel = lane & 15;
  const int qRow0 = qBase + wave * 16 + rsel;
  const int kend = qBase + 64;

  for (int kb = 0; kb < kend; kb += 32) {
#if HAVE_TDM
    if (wave == 0)
      tdm_load_tile_f16((unsigned)(unsigned long long)(void*)&sK[0][0],
                        Kp + (size_t)kb * HD, 32);
#endif
    for (int c0 = tid; c0 < 32 * 16; c0 += 128) {
      const int r = c0 >> 4, kc = (c0 & 15) << 3;
#if !HAVE_TDM
      *(v8h*)&sK[r][kc] = *(const v8h*)(Kp + (size_t)(kb + r) * HD + kc);
#endif
      v8h vv = *(const v8h*)(Vp + (size_t)(kb + r) * HD + kc);
#pragma unroll
      for (int d = 0; d < 8; ++d) sVt[kc + d][r] = vv[d];
      if (kb + 32 < kend)
        __builtin_prefetch(Vp + (size_t)(kb + 32 + r) * HD + kc, 0, 0);
    }
#if HAVE_TDM
    if (wave == 0) __builtin_amdgcn_s_wait_tensorcnt((short)0);
#endif
    __syncthreads();

    v8f s0 = {}, s1 = {};
#pragma unroll
    for (int kc = 0; kc < 4; ++kc) {
      v16h b0 = ldsB_frag(&sK[0][0], 136, 0, kc * 32, lane);
      v16h b1 = ldsB_frag(&sK[0][0], 136, 16, kc * 32, lane);
      s0 = wmma16(aQ[kc], b0, s0);
      s1 = wmma16(aQ[kc], b1, s1);
    }

    float p0[8], p1[8], alpha[8];
#pragma unroll
    for (int e = 0; e < 8; ++e) {
      const int q = qRow0 + e;
      float v0 = s0[e] * scale;
      float v1 = s1[e] * scale;
      if (kb + csel > q)      v0 = -1e30f;
      if (kb + 16 + csel > q) v1 = -1e30f;
      float mx = fmaxf(v0, v1);
      mx = fmaxf(mx, __shfl_xor(mx, 1, 32));
      mx = fmaxf(mx, __shfl_xor(mx, 2, 32));
      mx = fmaxf(mx, __shfl_xor(mx, 4, 32));
      mx = fmaxf(mx, __shfl_xor(mx, 8, 32));
      const float mnew = fmaxf(rmax[e], mx);
      alpha[e] = __expf(rmax[e] - mnew);
      rmax[e] = mnew;
      const float e0 = __expf(v0 - mnew);
      const float e1 = __expf(v1 - mnew);
      p0[e] = e0; p1[e] = e1;
      float ss = e0 + e1;
      ss += __shfl_xor(ss, 1, 32);
      ss += __shfl_xor(ss, 2, 32);
      ss += __shfl_xor(ss, 4, 32);
      ss += __shfl_xor(ss, 8, 32);
      rsum[e] = rsum[e] * alpha[e] + ss;
    }

#pragma unroll
    for (int t = 0; t < 8; ++t)
#pragma unroll
      for (int e = 0; e < 8; ++e) o[t][e] *= alpha[e];

    // Rotate P from C-layout to A-layout through wave-private LDS.
#pragma unroll
    for (int e = 0; e < 8; ++e) {
      sP[wave][e + rsel][csel]      = (_Float16)p0[e];
      sP[wave][e + rsel][16 + csel] = (_Float16)p1[e];
    }
    v16h aP = ldsA_frag(&sP[wave][0][0], 40, 0, 0, lane);
#pragma unroll
    for (int t = 0; t < 8; ++t) {
      v16h bv = ldsB_frag(&sVt[0][0], 40, t * 16, 0, lane);
      o[t] = wmma16(aP, bv, o[t]);
    }
    __syncthreads();
  }

  // Normalize and write f16 attention output, laid out (B, T, C).
#pragma unroll
  for (int t = 0; t < 8; ++t)
#pragma unroll
    for (int e = 0; e < 8; ++e) {
      const int q = qRow0 + e;
      const int dim = t * 16 + csel;
      Ao[((size_t)b * TT + q) * CC + h * HD + dim] = (_Float16)(o[t][e] / rsum[e]);
    }
}

// ---------------------------------------------------------------------------
// Kernel 4: output projection  out = attn @ Wo^T + bo  (fp32 output).
// ---------------------------------------------------------------------------
__global__ __launch_bounds__(128) void k_oproj(const _Float16* __restrict__ Ain,
                                               const float* __restrict__ Wo,
                                               const float* __restrict__ bo,
                                               float* __restrict__ out) {
  __shared__ __attribute__((aligned(16))) _Float16 sA[64][40];
  __shared__ __attribute__((aligned(16))) _Float16 sB[64][40];
  const int tid = threadIdx.x, lane = tid & 31, wave = tid >> 5;
  const int mBase = blockIdx.y * 64;
  const int nBase = blockIdx.x * 64;
  const int wm = (wave >> 1) * 32, wn = (wave & 1) * 32;
  v8f acc[2][2] = {};
  const int lr = tid >> 3, lk = (tid & 7) * 4;

  for (int kt = 0; kt < CC; kt += 32) {
#pragma unroll
    for (int c0 = tid; c0 < 256; c0 += 128) {
      const int r = c0 >> 2, kc = (c0 & 3) << 3;
      *(v8h*)&sA[r][kc] = *(const v8h*)(Ain + (size_t)(mBase + r) * CC + kt + kc);
    }
#pragma unroll
    for (int rg = 0; rg < 64; rg += 16) {
      const float* wrow = Wo + (size_t)(nBase + lr + rg) * CC + kt + lk;
      const float4 vb = *(const float4*)wrow;
      sB[lr + rg][lk + 0] = (_Float16)vb.x;
      sB[lr + rg][lk + 1] = (_Float16)vb.y;
      sB[lr + rg][lk + 2] = (_Float16)vb.z;
      sB[lr + rg][lk + 3] = (_Float16)vb.w;
      if (kt + 32 < CC) __builtin_prefetch(wrow + 32, 0, 0);
    }
    __syncthreads();
#pragma unroll
    for (int i = 0; i < 2; ++i) {
      v16h af = ldsA_frag(&sA[0][0], 40, wm + i * 16, 0, lane);
#pragma unroll
      for (int j = 0; j < 2; ++j) {
        v16h bf = ldsB_frag(&sB[0][0], 40, wn + j * 16, 0, lane);
        acc[i][j] = wmma16(af, bf, acc[i][j]);
      }
    }
    __syncthreads();
  }

  const int rsel = (lane >> 4) * 8, csel = lane & 15;
#pragma unroll
  for (int i = 0; i < 2; ++i)
#pragma unroll
    for (int j = 0; j < 2; ++j)
#pragma unroll
      for (int e = 0; e < 8; ++e) {
        const int m = mBase + wm + i * 16 + e + rsel;
        const int n = nBase + wn + j * 16 + csel;
        out[(size_t)m * CC + n] = acc[i][j][e] + bo[n];
      }
}

// ---------------------------------------------------------------------------
extern "C" void kernel_launch(void* const* d_in, const int* in_sizes, int n_in,
                              void* d_out, int out_size, void* d_ws, size_t ws_size,
                              hipStream_t stream) {
  (void)in_sizes; (void)n_in; (void)out_size; (void)ws_size;
  const float* x   = (const float*)d_in[0];
  const float* cs  = (const float*)d_in[1];
  const float* sn  = (const float*)d_in[2];
  const float* Wq  = (const float*)d_in[3];
  const float* Wkv = (const float*)d_in[4];
  const float* Wo  = (const float*)d_in[5];
  const float* bo  = (const float*)d_in[6];

  char* ws = (char*)d_ws;
  const size_t qBytes = (size_t)NB * NH * TT * HD * sizeof(_Float16);   // 32 MB
  const size_t kBytes = (size_t)NB * NKV * TT * HD * sizeof(_Float16);  //  8 MB
  _Float16* Qh = (_Float16*)ws;
  _Float16* Kh = (_Float16*)(ws + qBytes);
  _Float16* Vh = (_Float16*)(ws + qBytes + kBytes);
  _Float16* Ah = (_Float16*)(ws + qBytes + 2 * kBytes);                 // 32 MB

  dim3 blk(128);
  k_qkv<<<dim3(NTOT / 64, (NB * TT) / 64), blk, 0, stream>>>(x, Wq, Wkv, Qh, Kh, Vh);

  const long nrope = (long)NB * NH * TT * (HD / 2) + (long)NB * NKV * TT * (HD / 2);
  k_rope<<<dim3((unsigned)(nrope / 256)), dim3(256), 0, stream>>>(Qh, Kh, cs, sn);

  k_flash<<<dim3(TT / 64, NH, NB), blk, 0, stream>>>(Qh, Kh, Vh, Ah);

  k_oproj<<<dim3(CC / 64, (NB * TT) / 64), blk, 0, stream>>>(Ah, Wo, bo, (float*)d_out);
}
